// MultiHeadAttention_24781961298465
// MI455X (gfx1250) — matrix operand order repaired and checked by reading, not compiled, not run on hardware
//
#include <hip/hip_runtime.h>
#include <hip/hip_bf16.h>
#include <stddef.h>

// ---- problem constants (B,S,D,H from reference) ----
#define BB   2
#define SS   2048
#define DD   1024
#define HH   16
#define HD   64
#define NEGV (-100000.0f)

typedef unsigned short u16;
typedef __attribute__((ext_vector_type(16))) __bf16   v16bf;
typedef __attribute__((ext_vector_type(8)))  float    v8f;
typedef __attribute__((ext_vector_type(4)))  float    v4f;
typedef __attribute__((ext_vector_type(4)))  unsigned int v4u;
typedef __attribute__((ext_vector_type(2)))  unsigned int v2u;

// TDM descriptor group types (6-arg builtin form: g0,g1,g2,g3,g4,cpol)
typedef __attribute__((ext_vector_type(4))) unsigned int tdm4u;
typedef __attribute__((ext_vector_type(8))) int          tdm8i;
typedef __attribute__((ext_vector_type(4))) int          tdm4i;

#if defined(__has_builtin)
#if __has_builtin(__builtin_amdgcn_tensor_load_to_lds) && \
    __has_builtin(__builtin_amdgcn_s_wait_tensorcnt)
#define USE_TDM 1
#endif
#endif

union BF16x16 { v16bf v; v4u q[2]; unsigned int u[8]; u16 s[16]; };

__device__ __forceinline__ u16 f2bf(float f) {
    union { float f; unsigned int u; } x; x.f = f;
    unsigned int u = x.u;
    u += 0x7FFFu + ((u >> 16) & 1u);   // round-to-nearest-even
    return (u16)(u >> 16);
}

// pack two f32 -> packed bf16x2 (prefer HW packed convert)
__device__ __forceinline__ unsigned int pack2bf(float a, float b) {
#if defined(__has_builtin) && __has_builtin(__builtin_amdgcn_cvt_pk_bf16_f32)
    auto r = __builtin_amdgcn_cvt_pk_bf16_f32(a, b);
    unsigned int u = 0;
    __builtin_memcpy(&u, &r, sizeof(r) < 4 ? sizeof(r) : 4);
    return u;
#else
    return (unsigned)f2bf(a) | ((unsigned)f2bf(b) << 16);
#endif
}

// ---------------------------------------------------------------------------
// f32 -> bf16 conversion (weights)
// ---------------------------------------------------------------------------
__global__ __launch_bounds__(256)
void cvt_kernel(const float* __restrict__ src, u16* __restrict__ dst) {
    int i = (blockIdx.x * 256 + threadIdx.x) * 4;
    v4f x = *(const v4f*)(src + i);
    v2u p;
    p.x = pack2bf(x.x, x.y);
    p.y = pack2bf(x.z, x.w);
    *(v2u*)(dst + i) = p;
}

// ---------------------------------------------------------------------------
// GEMM: Y[m,n] = sum_k A[m,k] * W[n,k]   (Linear with W=[out,in])
// Block = 64(M) x 128(N), 8 waves, ping-pong LDS A tiles (1 barrier/k-step).
// mode 0: bf16 out, Q/K head-split [B,H,S,HD]
// mode 1: bf16 out, V transposed   [B,H,HD,S]
// mode 2: f32 out, plain [M,N]
// ---------------------------------------------------------------------------
__global__ __launch_bounds__(256)
void gemm_kernel(const float* __restrict__ Af32, const u16* __restrict__ Abf,
                 const u16* __restrict__ Wbf,
                 u16* __restrict__ Yb, float* __restrict__ Yf, const int mode)
{
    __shared__ alignas(16) u16 As[2][64][32];

    const int t    = threadIdx.x;
    const int lane = t & 31;
    const int w    = t >> 5;
    const int lh   = lane >> 4;   // wave half (0/1)
    const int ll   = lane & 15;
    const int m0   = blockIdx.x * 64;
    const int n0   = blockIdx.y * 128 + w * 16;

    v8f acc[4] = {};

    const int arow = t >> 2;            // 0..63
    const int aoff = (t & 3) * 8;       // 0,8,16,24

    auto load_tile = [&](int k0) -> v4u {
        v4u pk;
        if (Af32) {
            const float* ap = Af32 + (size_t)(m0 + arow) * DD + k0 + aoff;
            v4f x0 = *(const v4f*)(ap);
            v4f x1 = *(const v4f*)(ap + 4);
            pk.x = pack2bf(x0.x, x0.y);
            pk.y = pack2bf(x0.z, x0.w);
            pk.z = pack2bf(x1.x, x1.y);
            pk.w = pack2bf(x1.z, x1.w);
        } else {
            pk = *(const v4u*)(Abf + (size_t)(m0 + arow) * DD + k0 + aoff);
        }
        return pk;
    };

    // prologue: stage tile 0
    {
        v4u c = load_tile(0);
        *(v4u*)&As[0][arow][aoff] = c;
    }
    __syncthreads();

    int buf = 0;
    for (int k0 = 0; k0 < DD; k0 += 32) {
        const bool last = (k0 + 32 >= DD);
        v4u nxt = {};
        if (!last) nxt = load_tile(k0 + 32);   // issue global loads early

        // B fragment: k = lh*8 + i, 16 + lh*8 + i
        BF16x16 bfrag;
        const u16* wrow = Wbf + (size_t)(n0 + ll) * DD + k0;
        bfrag.q[0] = *(const v4u*)(wrow + lh * 8);
        bfrag.q[1] = *(const v4u*)(wrow + 16 + lh * 8);

        #pragma unroll
        for (int ms = 0; ms < 4; ++ms) {
            BF16x16 af;
            const u16* ar = &As[buf][ms * 16 + ll][0];
            af.q[0] = *(const v4u*)(ar + lh * 8);          // k = lh*8 ..
            af.q[1] = *(const v4u*)(ar + 16 + lh * 8);     // k = 16+lh*8 ..
            acc[ms] = __builtin_amdgcn_wmma_f32_16x16x32_bf16(
                false, af.v, false, bfrag.v, (short)0, acc[ms], false, false);
        }

        if (!last) *(v4u*)&As[buf ^ 1][arow][aoff] = nxt;
        __syncthreads();
        buf ^= 1;
    }

    // ---- epilogue: C layout m = r + 8*lh, n = ll ----
    #pragma unroll
    for (int ms = 0; ms < 4; ++ms) {
        #pragma unroll
        for (int r = 0; r < 8; ++r) {
            int m = m0 + ms * 16 + r + 8 * lh;
            int n = n0 + ll;
            float val = acc[ms][r];
            if (mode == 2) {
                Yf[(size_t)m * DD + n] = val;
            } else {
                int b = m >> 11, s = m & (SS - 1);
                int h = n >> 6,  hd = n & 63;
                size_t idx = (mode == 0)
                    ? ((size_t)(b * HH + h) * SS + s) * HD + hd
                    : ((size_t)(b * HH + h) * HD + hd) * SS + s;
                Yb[idx] = f2bf(val);
            }
        }
    }
}

// ---------------------------------------------------------------------------
// TDM issue: K chunk (contiguous 2048 elems) + V chunk (2D 32x64, stride S)
// D# built per CDNA5 ISA 8.3/8.4 bit layout; data_size=2B; type=2 ("image").
// ---------------------------------------------------------------------------
#ifdef USE_TDM
__device__ __forceinline__ void tdm_issue_kv(const u16* gK, unsigned ldsK,
                                             const u16* gV, unsigned ldsV)
{
    tdm4i gz4 = {0, 0, 0, 0};
    tdm8i gz8 = {0, 0, 0, 0, 0, 0, 0, 0};
    tdm4u g0; tdm8i g1;

    // --- K: 1D contiguous, tile_dim0 = 2048 elements (4KB) ---
    unsigned long long ga = (unsigned long long)(uintptr_t)gK;
    g0.x = 1u;                                            // count=1 (valid)
    g0.y = ldsK;                                          // lds_addr (bytes)
    g0.z = (unsigned)ga;                                  // global_addr lo
    g0.w = (unsigned)((ga >> 32) & 0x1FFFFFFu) | 0x80000000u;  // hi + type=2
    g1[0] = (int)(1u << 16);                              // data_size = 2B
    g1[1] = (int)((2048u & 0xFFFFu) << 16);               // tensor_dim0 lo
    g1[2] = (int)((2048u >> 16) | (1u << 16));            // dim0 hi | dim1 lo=1
    g1[3] = (int)(2048u << 16);                           // dim1 hi | tile_dim0
    g1[4] = 0;                                            // tile_dim1/2 unused
    g1[5] = 2048;                                         // stride0 lo
    g1[6] = 0;
    g1[7] = 0;
    __builtin_amdgcn_tensor_load_to_lds(g0, g1, gz4, gz4, gz8, 0);

    // --- V: 2D tile 32(x) x 64(rows), row stride = S elements ---
    unsigned long long gv = (unsigned long long)(uintptr_t)gV;
    g0.x = 1u;
    g0.y = ldsV;
    g0.z = (unsigned)gv;
    g0.w = (unsigned)((gv >> 32) & 0x1FFFFFFu) | 0x80000000u;
    g1[0] = (int)(1u << 16);                              // data_size = 2B
    g1[1] = (int)(32u << 16);                             // tensor_dim0 = 32
    g1[2] = (int)(64u << 16);                             // | tensor_dim1 = 64
    g1[3] = (int)(32u << 16);                             // | tile_dim0 = 32
    g1[4] = 64;                                           // tile_dim1 = 64
    g1[5] = (int)SS;                                      // stride0 = 2048
    g1[6] = 0;
    g1[7] = 0;
    __builtin_amdgcn_tensor_load_to_lds(g0, g1, gz4, gz4, gz8, 0);
}
#endif

// ---------------------------------------------------------------------------
// Flash-style attention. Grid (S/64, B*H), block 128 (4 waves).
// Each wave owns a 16-query tile; K/V chunks staged in LDS (TDM async,
// double-buffered, prefetching chunk n+1 while computing chunk n).
// ---------------------------------------------------------------------------
__global__ __launch_bounds__(128)
void attn_kernel(const u16* __restrict__ Qb, const u16* __restrict__ Kb,
                 const u16* __restrict__ Vt, const int* __restrict__ maskp,
                 u16* __restrict__ Oac)
{
    __shared__ alignas(16) u16 Ks[2][32][64];      // [buf][key][hd]
    __shared__ alignas(16) u16 Vs[2][64][32];      // [buf][hd][key]
    __shared__ alignas(16) u16 Ps[4][16][32];      // per-wave P tile [m][k]

    const int t    = threadIdx.x;
    const int lane = t & 31;
    const int w    = t >> 5;
    const int lh   = lane >> 4;
    const int ll   = lane & 15;
    const int bh   = blockIdx.y;
    const int b    = bh >> 4;
    const int h    = bh & 15;
    const int q0   = (blockIdx.x * 4 + w) * 16;

    const u16* Kbase = Kb + (size_t)bh * SS * HD;
    const u16* Vbase = Vt + (size_t)bh * HD * SS;

    // Q A-fragments for 2 k-steps over HD=64 (kept in registers)
    BF16x16 qa[2];
    {
        const u16* qrow = Qb + ((size_t)bh * SS + q0 + ll) * HD;
        #pragma unroll
        for (int ks = 0; ks < 2; ++ks) {
            qa[ks].q[0] = *(const v4u*)(qrow + ks * 32 + lh * 8);
            qa[ks].q[1] = *(const v4u*)(qrow + ks * 32 + 16 + lh * 8);
        }
    }

    float mrow[8], lrow[8];
    v8f o[4] = {};
    #pragma unroll
    for (int r = 0; r < 8; ++r) { mrow[r] = -3.0e38f; lrow[r] = 0.0f; }

    int buf = 0;
#ifdef USE_TDM
    if (w == 0) {
        tdm_issue_kv(Kbase, (unsigned)(uintptr_t)&Ks[0][0][0],
                     Vbase, (unsigned)(uintptr_t)&Vs[0][0][0]);
    }
#endif

    for (int kc = 0; kc < SS; kc += 32) {
#ifdef USE_TDM
        if (w == 0) {
            if (kc + 32 < SS) {
                // prefetch next chunk into the other buffer, then wait for
                // the 2 older (in-order) tensor loads of the current chunk
                tdm_issue_kv(Kbase + (size_t)(kc + 32) * HD,
                             (unsigned)(uintptr_t)&Ks[buf ^ 1][0][0],
                             Vbase + (kc + 32),
                             (unsigned)(uintptr_t)&Vs[buf ^ 1][0][0]);
                __builtin_amdgcn_s_wait_tensorcnt((short)2);
            } else {
                __builtin_amdgcn_s_wait_tensorcnt((short)0);
            }
        }
        __syncthreads();   // current chunk visible to all waves
#else
        {   // cooperative fallback: K chunk [32 keys][64 hd]
            int row = t >> 2, off = (t & 3) * 16;
            const u16* kp = Kbase + (size_t)(kc + row) * HD + off;
            *(v4u*)&Ks[buf][row][off]     = *(const v4u*)(kp);
            *(v4u*)&Ks[buf][row][off + 8] = *(const v4u*)(kp + 8);
            int hd = t >> 1, o2 = (t & 1) * 16;
            const u16* vp = Vbase + (size_t)hd * SS + kc + o2;
            *(v4u*)&Vs[buf][hd][o2]     = *(const v4u*)(vp);
            *(v4u*)&Vs[buf][hd][o2 + 8] = *(const v4u*)(vp + 8);
        }
        __syncthreads();
#endif

        // ---- scores: two 16x16 tiles (keys kc..+15, kc+16..+31) ----
        v8f sc[2];
        #pragma unroll
        for (int sub = 0; sub < 2; ++sub) {
            BF16x16 kb0, kb1;
            const u16* krow = &Ks[buf][sub * 16 + ll][0];
            kb0.q[0] = *(const v4u*)(krow + lh * 8);
            kb0.q[1] = *(const v4u*)(krow + 16 + lh * 8);
            kb1.q[0] = *(const v4u*)(krow + 32 + lh * 8);
            kb1.q[1] = *(const v4u*)(krow + 48 + lh * 8);
            v8f z = {};
            z = __builtin_amdgcn_wmma_f32_16x16x32_bf16(
                false, qa[0].v, false, kb0.v, (short)0, z, false, false);
            z = __builtin_amdgcn_wmma_f32_16x16x32_bf16(
                false, qa[1].v, false, kb1.v, (short)0, z, false, false);
            sc[sub] = z;
        }

        // ---- mask + scale + online softmax (rows spread over 16 lanes) ----
        int mv0 = maskp[b * SS + kc + ll];
        int mv1 = maskp[b * SS + kc + 16 + ll];
        float p0[8], p1[8];
        #pragma unroll
        for (int r = 0; r < 8; ++r) {
            p0[r] = mv0 ? sc[0][r] * 0.125f : NEGV;
            p1[r] = mv1 ? sc[1][r] * 0.125f : NEGV;
        }
        #pragma unroll
        for (int r = 0; r < 8; ++r) {
            float tmax = fmaxf(p0[r], p1[r]);
            #pragma unroll
            for (int i = 1; i < 16; i <<= 1)
                tmax = fmaxf(tmax, __shfl_xor(tmax, i, 32));
            float nm = fmaxf(mrow[r], tmax);
            float al = __expf(mrow[r] - nm);
            float e0 = mv0 ? __expf(p0[r] - nm) : 0.0f;
            float e1 = mv1 ? __expf(p1[r] - nm) : 0.0f;
            p0[r] = e0; p1[r] = e1;
            float rs = e0 + e1;
            #pragma unroll
            for (int i = 1; i < 16; i <<= 1)
                rs += __shfl_xor(rs, i, 32);
            lrow[r] = lrow[r] * al + rs;
            mrow[r] = nm;
            #pragma unroll
            for (int j = 0; j < 4; ++j) o[j][r] *= al;
        }

        // ---- C-layout -> A-layout for P via LDS ----
        #pragma unroll
        for (int r = 0; r < 8; ++r) {
            Ps[w][r + 8 * lh][ll]      = f2bf(p0[r]);
            Ps[w][r + 8 * lh][16 + ll] = f2bf(p1[r]);
        }
        __syncthreads();
        BF16x16 pa;
        {
            const u16* prow = &Ps[w][ll][0];
            pa.q[0] = *(const v4u*)(prow + lh * 8);
            pa.q[1] = *(const v4u*)(prow + 16 + lh * 8);
        }

        // ---- O += P(16x32) x V(32x64): 4 N-subtiles of hd ----
        #pragma unroll
        for (int j = 0; j < 4; ++j) {
            BF16x16 vb;
            const u16* vrow = &Vs[buf][j * 16 + ll][0];
            vb.q[0] = *(const v4u*)(vrow + lh * 8);
            vb.q[1] = *(const v4u*)(vrow + 16 + lh * 8);
            o[j] = __builtin_amdgcn_wmma_f32_16x16x32_bf16(
                false, pa.v, false, vb.v, (short)0, o[j], false, false);
        }

        __syncthreads();   // all reads of this buffer done
        buf ^= 1;
    }

    // ---- epilogue: normalize, store bf16 into [B,S, h*64+hd] ----
    #pragma unroll
    for (int j = 0; j < 4; ++j) {
        #pragma unroll
        for (int r = 0; r < 8; ++r) {
            float denom = lrow[r];
            float val = (denom > 0.0f) ? o[j][r] / denom : 0.0f;
            int s   = q0 + r + 8 * lh;
            int col = h * HD + j * 16 + ll;
            Oac[((size_t)b * SS + s) * DD + col] = f2bf(val);
        }
    }
}

// ---------------------------------------------------------------------------
// Host launcher
// ---------------------------------------------------------------------------
extern "C" void kernel_launch(void* const* d_in, const int* in_sizes, int n_in,
                              void* d_out, int out_size, void* d_ws, size_t ws_size,
                              hipStream_t stream)
{
    const float* q    = (const float*)d_in[0];
    const float* k    = (const float*)d_in[1];
    const float* v    = (const float*)d_in[2];
    const int*   mask = (const int*)  d_in[3];
    const float* Wq   = (const float*)d_in[4];
    const float* Wk   = (const float*)d_in[5];
    const float* Wv   = (const float*)d_in[6];
    const float* Wo   = (const float*)d_in[7];
    float* out = (float*)d_out;

    const size_t DDsz = (size_t)DD * DD;        // 1M
    const size_t BSD  = (size_t)BB * SS * DD;   // 4M

    u16* ws16 = (u16*)d_ws;
    u16* Wqb = ws16;
    u16* Wkb = Wqb + DDsz;
    u16* Wvb = Wkb + DDsz;
    u16* Wob = Wvb + DDsz;
    u16* Qb  = Wob + DDsz;
    u16* Kb  = Qb + BSD;
    u16* Vt  = Kb + BSD;
    u16* Oac = Vt + BSD;

    // weight conversion (f32 -> bf16)
    dim3 cgrid((unsigned)(DDsz / (256 * 4)));
    cvt_kernel<<<cgrid, 256, 0, stream>>>(Wq, Wqb);
    cvt_kernel<<<cgrid, 256, 0, stream>>>(Wk, Wkb);
    cvt_kernel<<<cgrid, 256, 0, stream>>>(Wv, Wvb);
    cvt_kernel<<<cgrid, 256, 0, stream>>>(Wo, Wob);

    // projections
    dim3 ggrid((BB * SS) / 64, DD / 128);       // (64, 8)
    gemm_kernel<<<ggrid, 256, 0, stream>>>(q, nullptr, Wqb, Qb, nullptr, 0);
    gemm_kernel<<<ggrid, 256, 0, stream>>>(k, nullptr, Wkb, Kb, nullptr, 0);
    gemm_kernel<<<ggrid, 256, 0, stream>>>(v, nullptr, Wvb, Vt, nullptr, 1);

    // attention
    dim3 agrid(SS / 64, BB * HH);               // (32, 32)
    attn_kernel<<<agrid, 128, 0, stream>>>(Qb, Kb, Vt, mask, Oac);

    // output projection (bf16 A, f32 out)
    gemm_kernel<<<ggrid, 256, 0, stream>>>(nullptr, Oac, Wob, nullptr, out, 2);
}
